// ChildSumTreeLSTMOp_12644383719723
// MI455X (gfx1250) — matrix operand order repaired and checked by reading, not compiled, not run on hardware
//
#include <hip/hip_runtime.h>
#include <hip/hip_bf16.h>

typedef __attribute__((ext_vector_type(16))) _Float16 v16h;
typedef __attribute__((ext_vector_type(8)))  float    v8f;

#define D 256
#define KB 8          // 256 / 32 K-steps
#define WPB 4         // waves per block
#define TPB 128       // threads per block

// ---------- helpers ----------
static __device__ __forceinline__ float fast_rcp(float x) {
    return __builtin_amdgcn_rcpf(x);      // v_rcp_f32 (TRANS, co-executes with VALU)
}
static __device__ __forceinline__ float sigm(float x) {
    return fast_rcp(1.0f + __expf(-x));
}
static __device__ __forceinline__ float fast_tanh(float x) {
    // tanh(x) = 1 - 2/(e^(2x)+1): one v_exp_f32 + one v_rcp_f32 + fma
    x = fminf(15.0f, fmaxf(-15.0f, x));
    return 1.0f - 2.0f * fast_rcp(__expf(2.0f * x) + 1.0f);
}

// A fragment (16x32 f16): lane half h holds K = h*8..h*8+7 in elems 0..7,
// K = 16+h*8.. in elems 8..15. p points at row_start + kb*32 (f32 source).
static __device__ __forceinline__ v16h load_fragA(const float* __restrict__ p, int half) {
    v16h v;
    const float* p0 = p + half * 8;
    const float* p1 = p + 16 + half * 8;
#pragma unroll
    for (int i = 0; i < 8; i++) v[i] = (_Float16)p0[i];
#pragma unroll
    for (int i = 0; i < 8; i++) v[8 + i] = (_Float16)p1[i];
    return v;
}

// B fragment (32x16 f16): lane (l&15) = column, K = half*8.. in elems 0..7, K = 16+half*8.. in elems 8..15.
static __device__ __forceinline__ v16h load_fragB_lds(const _Float16* __restrict__ p, int half) {
    v16h v;
    const _Float16* q0 = p + half * 8;
    const _Float16* q1 = p + 16 + half * 8;
#pragma unroll
    for (int i = 0; i < 8; i++) v[i] = q0[i];
#pragma unroll
    for (int i = 0; i < 8; i++) v[8 + i] = q1[i];
    return v;
}

static __device__ __forceinline__ v8f wmma_f16(v16h a, v16h b, v8f c) {
    return __builtin_amdgcn_wmma_f32_16x16x32_f16(false, a, false, b, (short)0, c, false, false);
}

// ---------- kernel 0: zero workspace ----------
__global__ void zero_ws_kernel(float4* __restrict__ p, long n4) {
    long i = (long)blockIdx.x * blockDim.x + threadIdx.x;
    if (i < n4) p[i] = float4{0.f, 0.f, 0.f, 0.f};
}

// ---------- kernel 1: fused edge stage ----------
// Per 16-edge tile (one wave):
//   h_sum[dst]  += h[src]                  (fused into A-fragment load, f32 atomics)
//   f            = sigmoid(h[src]@Uf^T+b)  (f16 WMMA, f32 accum, Uf slices in LDS)
//   c_agg[dst]  += f * c[src]              (branchless zeroed atomics for tail)
__global__ void __launch_bounds__(TPB)
edge_gemm_kernel(const float* __restrict__ h, const float* __restrict__ c,
                 const int* __restrict__ src, const int* __restrict__ dst,
                 const float* __restrict__ Uf, const float* __restrict__ Ufb,
                 float* __restrict__ hsum, float* __restrict__ cagg, int E) {
    __shared__ _Float16 sB[16 * D];   // 8KB: current 16-row slice of U_f in f16

    const int tid  = threadIdx.x;
    const int wave = tid >> 5;
    const int lane = tid & 31;
    const int half = lane >> 4;
    const int lm   = lane & 15;
    const int tile = blockIdx.x * WPB + wave;
    const int e0   = tile * 16;

    // A-fragment row for this lane: edge e0+lm (clamped); fused h_sum scatter.
    const bool okA = (e0 + lm) < E;
    const int  eA  = min(e0 + lm, E - 1);
    const float* arow = h    + (size_t)src[eA] * D;
    float*       hout = hsum + (size_t)dst[eA] * D;

    v16h a[KB];
#pragma unroll
    for (int kb = 0; kb < KB; kb++) {
        const int o0 = kb * 32 + half * 8;
        const int o1 = kb * 32 + 16 + half * 8;
        float f0[8], f1[8];
#pragma unroll
        for (int i = 0; i < 8; i++) f0[i] = arow[o0 + i];
#pragma unroll
        for (int i = 0; i < 8; i++) f1[i] = arow[o1 + i];
        v16h v;
#pragma unroll
        for (int i = 0; i < 8; i++) v[i]     = (_Float16)f0[i];
#pragma unroll
        for (int i = 0; i < 8; i++) v[8 + i] = (_Float16)f1[i];
        a[kb] = v;
        // lanes lm and lm+16 disjointly cover the full 256-wide row
#pragma unroll
        for (int i = 0; i < 8; i++) atomicAdd(hout + o0 + i, okA ? f0[i] : 0.0f);
#pragma unroll
        for (int i = 0; i < 8; i++) atomicAdd(hout + o1 + i, okA ? f1[i] : 0.0f);
    }

    // per-lane edge bookkeeping for the C/D rows this lane owns
    int s_m[8], d_m[8];
    bool ok_m[8];
#pragma unroll
    for (int r = 0; r < 8; r++) {
        int e = e0 + r + 8 * half;
        ok_m[r] = e < E;
        e = min(e, E - 1);
        s_m[r] = src[e];
        d_m[r] = dst[e];
    }

    for (int nt = 0; nt < 16; nt++) {
        __syncthreads();
        // stage U_f rows [nt*16, nt*16+16) as f16 into LDS (cooperative)
        for (int i4 = tid; i4 < (16 * D) / 4; i4 += TPB) {
            int row = i4 >> 6;            // (i4*4)/256
            int col = (i4 & 63) * 4;
            const float4 v = *(const float4*)(Uf + (size_t)(nt * 16 + row) * D + col);
            _Float16* o = sB + row * D + col;
            o[0] = (_Float16)v.x; o[1] = (_Float16)v.y;
            o[2] = (_Float16)v.z; o[3] = (_Float16)v.w;
        }
        __syncthreads();

        v8f acc = {};
#pragma unroll
        for (int kb = 0; kb < KB; kb++) {
            v16h b = load_fragB_lds(sB + lm * D + kb * 32, half);
            acc = wmma_f16(a[kb], b, acc);
        }

        const int j = nt * 16 + lm;
        const float bj = Ufb[j];
#pragma unroll
        for (int r = 0; r < 8; r++) {
            float f   = sigm(acc[r] + bj);
            float cs  = c[(size_t)s_m[r] * D + j];
            float val = ok_m[r] ? (f * cs) : 0.0f;   // branchless tail handling
            atomicAdd(cagg + (size_t)d_m[r] * D + j, val);
        }
    }
}

// ---------- kernel 2: iou_t = x@W^T + hsum@U^T + b; gates; write h_new, c_new ----------
__global__ void __launch_bounds__(TPB)
node_gemm_kernel(const float* __restrict__ x, const float* __restrict__ hsum,
                 const float* __restrict__ W, const float* __restrict__ U,
                 const float* __restrict__ biou, const float* __restrict__ cagg,
                 float* __restrict__ out, int N_) {
    __shared__ _Float16 sB[96 * D];   // 48KB: 6 x 16-row weight slices (i/o/u x {W,U})

    const int tid  = threadIdx.x;
    const int wave = tid >> 5;
    const int lane = tid & 31;
    const int half = lane >> 4;
    const int lm   = lane & 15;
    const int tile = blockIdx.x * WPB + wave;
    const int n0   = tile * 16;
    const bool fullTile = (n0 + 16) <= N_;

    const int arow = min(n0 + lm, N_ - 1);
    const float* xr = x    + (size_t)arow * D;
    const float* hr = hsum + (size_t)arow * D;
    v16h ax[KB], ah[KB];
#pragma unroll
    for (int kb = 0; kb < KB; kb++) {
        ax[kb] = load_fragA(xr + kb * 32, half);
        ah[kb] = load_fragA(hr + kb * 32, half);
    }

    for (int t = 0; t < 16; t++) {
        __syncthreads();
        // stage 96 rows: g=0..2 -> W rows g*256 + t*16 + lr ; g=3..5 -> U rows (g-3)*256 + t*16 + lr
        for (int i4 = tid; i4 < (96 * D) / 4; i4 += TPB) {
            int row = i4 >> 6;
            int col = (i4 & 63) * 4;
            int g = row >> 4, lr = row & 15;
            const float* srcm = (g < 3) ? (W + (size_t)(g * 256 + t * 16 + lr) * D)
                                        : (U + (size_t)((g - 3) * 256 + t * 16 + lr) * D);
            const float4 v = *(const float4*)(srcm + col);
            _Float16* o = sB + row * D + col;
            o[0] = (_Float16)v.x; o[1] = (_Float16)v.y;
            o[2] = (_Float16)v.z; o[3] = (_Float16)v.w;
        }
        __syncthreads();

        v8f ai = {}, ao = {}, au = {};
#pragma unroll
        for (int kb = 0; kb < KB; kb++) {
            v16h b;
            b = load_fragB_lds(sB + (0 * 16 + lm) * D + kb * 32, half); ai = wmma_f16(ax[kb], b, ai);
            b = load_fragB_lds(sB + (3 * 16 + lm) * D + kb * 32, half); ai = wmma_f16(ah[kb], b, ai);
            b = load_fragB_lds(sB + (1 * 16 + lm) * D + kb * 32, half); ao = wmma_f16(ax[kb], b, ao);
            b = load_fragB_lds(sB + (4 * 16 + lm) * D + kb * 32, half); ao = wmma_f16(ah[kb], b, ao);
            b = load_fragB_lds(sB + (2 * 16 + lm) * D + kb * 32, half); au = wmma_f16(ax[kb], b, au);
            b = load_fragB_lds(sB + (5 * 16 + lm) * D + kb * 32, half); au = wmma_f16(ah[kb], b, au);
        }

        const int j = t * 16 + lm;
        const float bi = biou[j], bo = biou[256 + j], bu = biou[512 + j];
        if (fullTile) {
            // wave-uniform fast path: no per-element exec juggling
#pragma unroll
            for (int r = 0; r < 8; r++) {
                int n = n0 + r + 8 * half;
                float ca = cagg[(size_t)n * D + j];
                float iv = sigm(ai[r] + bi);
                float ov = sigm(ao[r] + bo);
                float uv = fast_tanh(au[r] + bu);
                float cn = iv * uv + ca;
                float hn = ov * fast_tanh(cn);
                out[(size_t)n * D + j]                  = hn;   // h_new
                out[(size_t)N_ * D + (size_t)n * D + j] = cn;   // c_new
            }
        } else {
#pragma unroll
            for (int r = 0; r < 8; r++) {
                int n = n0 + r + 8 * half;
                if (n < N_) {
                    float ca = cagg[(size_t)n * D + j];
                    float iv = sigm(ai[r] + bi);
                    float ov = sigm(ao[r] + bo);
                    float uv = fast_tanh(au[r] + bu);
                    float cn = iv * uv + ca;
                    float hn = ov * fast_tanh(cn);
                    out[(size_t)n * D + j]                  = hn;
                    out[(size_t)N_ * D + (size_t)n * D + j] = cn;
                }
            }
        }
    }
}

// ---------- launch ----------
extern "C" void kernel_launch(void* const* d_in, const int* in_sizes, int n_in,
                              void* d_out, int out_size, void* d_ws, size_t ws_size,
                              hipStream_t stream) {
    const float* x     = (const float*)d_in[0];
    const float* h     = (const float*)d_in[1];
    const float* c     = (const float*)d_in[2];
    const int*   src   = (const int*)d_in[3];
    const int*   dst   = (const int*)d_in[4];
    const float* Wiouf = (const float*)d_in[5];
    const float* Uiou  = (const float*)d_in[6];
    const float* biou  = (const float*)d_in[7];
    const float* UfW   = (const float*)d_in[8];
    const float* Ufb   = (const float*)d_in[9];
    float* out = (float*)d_out;

    const int N = in_sizes[0] / D;
    const int E = in_sizes[3];

    float* hsum = (float*)d_ws;
    float* cagg = hsum + (size_t)N * D;

    // 0) zero h_sum and c_agg (must happen each call: atomics accumulate)
    long n4 = (2L * N * D) / 4;
    long zb = (n4 + 255) / 256;
    zero_ws_kernel<<<(unsigned)zb, 256, 0, stream>>>((float4*)d_ws, n4);

    // 1) fused edge stage: h_sum scatter + f GEMM + gated c_agg scatter
    int etiles  = (E + 15) / 16;
    int eblocks = (etiles + WPB - 1) / WPB;
    edge_gemm_kernel<<<eblocks, TPB, 0, stream>>>(h, c, src, dst, UfW, Ufb, hsum, cagg, E);

    // 2) node GEMMs + gates + outputs
    int ntiles  = (N + 15) / 16;
    int nblocks = (ntiles + WPB - 1) / WPB;
    node_gemm_kernel<<<nblocks, TPB, 0, stream>>>(x, hsum, Wiouf, Uiou, biou, cagg, out, N);
}
